// MultiHeadAttention_89824946028703
// MI455X (gfx1250) — matrix operand order repaired and checked by reading, not compiled, not run on hardware
//
#include <hip/hip_runtime.h>

// ---------------------------------------------------------------------------
// MultiHeadAttention for MI455X (gfx1250, wave32, WMMA f32_16x16x32_f16)
//   B=8, S=1024, D_MODEL=1024, H=16, dk=dv=64
// d_out = out[8,1024,1024] f32  ||  attn[8,16,1024,1024] f32
// ---------------------------------------------------------------------------

typedef __attribute__((ext_vector_type(16))) _Float16 v16h;
typedef __attribute__((ext_vector_type(8)))  float    v8f;

#define BATCH 8
#define SEQ   1024
#define DM    1024
#define NH    16
#define DK    64

#ifndef __has_builtin
#define __has_builtin(x) 0
#endif
#if __has_builtin(__builtin_amdgcn_global_load_async_to_lds_b128) && \
    __has_builtin(__builtin_amdgcn_s_wait_asynccnt)
#define USE_ASYNC_LDS 1
#else
#define USE_ASYNC_LDS 0
#endif

#if USE_ASYNC_LDS
typedef int v4i_t __attribute__((ext_vector_type(4)));
typedef __attribute__((address_space(1))) v4i_t g_v4i;  // global int4
typedef __attribute__((address_space(3))) v4i_t l_v4i;  // LDS int4
#endif

// 16-byte global -> LDS copy.  On CDNA5 use the async-DMA path (ASYNCcnt);
// LDS flat pointers carry the LDS offset in their low 32 bits (ISA 10.2),
// flat global addresses are identity, so integer casts build the AS-typed
// pointers without addrspacecasts.
__device__ __forceinline__ void copy16_g2l(const _Float16* g, _Float16* l) {
#if USE_ASYNC_LDS
  __builtin_amdgcn_global_load_async_to_lds_b128(
      (g_v4i*)(unsigned long long)g,
      (l_v4i*)(unsigned int)(unsigned long long)l, 0, 0);
#else
  *(uint4*)l = *(const uint4*)g;
#endif
}
__device__ __forceinline__ void async_wait() {
#if USE_ASYNC_LDS
  __builtin_amdgcn_s_wait_asynccnt(0);
#endif
}

__device__ __forceinline__ v8f wmma_f16(v16h a, v16h b, v8f c) {
  return __builtin_amdgcn_wmma_f32_16x16x32_f16(
      /*neg_a=*/false, a, /*neg_b=*/false, b,
      /*c_mod=*/(short)0, c, /*reuse_a=*/false, /*reuse_b=*/false);
}

// A fragment: 16x32 (MxK) f16.  lane: m=lane&15, half=lane>>4
// VGPR v holds K = (v>=4?16:0) + half*8 + (v&3)*2 , +1   (ISA 7.12.2)
// source: row-major LDS tile, p[m*ld + k]
__device__ __forceinline__ v16h load_frag_a(const _Float16* p, int ld) {
  const int lane = threadIdx.x & 31;
  const int m = lane & 15, hf = lane >> 4;
  v16h a;
#pragma unroll
  for (int v = 0; v < 8; ++v) {
    const int k = ((v & 4) << 2) + hf * 8 + (v & 3) * 2;
    a[2 * v]     = p[m * ld + k];
    a[2 * v + 1] = p[m * ld + k + 1];
  }
  return a;
}

// B fragment: 32x16 (KxN) f16.  lane: n=lane&15, half=lane>>4
// VGPR v holds K = (v>=4?16:0) + half*8 + (v&3)*2 , +1
// source is "column-contiguous": B[k][n] = p[n*ld + k]
__device__ __forceinline__ v16h load_frag_b(const _Float16* p, int ld) {
  const int lane = threadIdx.x & 31;
  const int n = lane & 15, hf = lane >> 4;
  v16h b;
#pragma unroll
  for (int v = 0; v < 8; ++v) {
    const int k = ((v & 4) << 2) + hf * 8 + (v & 3) * 2;
    b[2 * v]     = p[n * ld + k];
    b[2 * v + 1] = p[n * ld + k + 1];
  }
  return b;
}
// C/D fragment (f32, 8 VGPRs): lane n = lane&15, rows m = (lane>>4)*8 + r.

// ---------------------------------------------------------------------------
// Kernel 1: QKV projections.  X[8192,1024](f32) @ W[1024,1024](f32) -> f16
// stored as [B,H,S,64].  grid (128, 16, 3), block 128 (4 waves), 64x64 tile.
// ---------------------------------------------------------------------------
__global__ __launch_bounds__(128) void qkv_proj_kernel(
    const float* __restrict__ Xq, const float* __restrict__ Xk,
    const float* __restrict__ Xv, const float* __restrict__ Wq,
    const float* __restrict__ Wk, const float* __restrict__ Wv,
    _Float16* __restrict__ Qh, _Float16* __restrict__ Kh,
    _Float16* __restrict__ Vh) {
  const int which = blockIdx.z;
  const float* X = (which == 0) ? Xq : (which == 1) ? Xk : Xv;
  const float* W = (which == 0) ? Wq : (which == 1) ? Wk : Wv;
  _Float16* O   = (which == 0) ? Qh : (which == 1) ? Kh : Vh;

  __shared__ _Float16 Xs[64 * 36];   // 64 rows x 32 k (ld 36, pad for banks)
  __shared__ _Float16 Wts[64 * 36];  // 64 n x 32 k (W transposed)

  const int t = threadIdx.x;
  const int wave = t >> 5;
  const int row0 = blockIdx.x * 64;
  const int col0 = blockIdx.y * 64;

  v8f acc[4] = {};

  for (int kt = 0; kt < DM; kt += 32) {
    __syncthreads();
    {  // X tile 64x32 f32 -> f16
      const int r = t >> 3, cg = (t & 7) * 4;
#pragma unroll
      for (int i = 0; i < 4; ++i) {
        const int row = r + i * 16;
        const float4 x =
            *(const float4*)(X + (size_t)(row0 + row) * DM + kt + cg);
        _Float16* dst = &Xs[row * 36 + cg];
        dst[0] = (_Float16)x.x; dst[1] = (_Float16)x.y;
        dst[2] = (_Float16)x.z; dst[3] = (_Float16)x.w;
      }
    }
    {  // W tile 32x64 f32 -> f16, transposed into Wts[n][k]
      const int r = t >> 4, cg = (t & 15) * 4;
#pragma unroll
      for (int i = 0; i < 4; ++i) {
        const int k = r + i * 8;
        const float4 w =
            *(const float4*)(W + (size_t)(kt + k) * DM + col0 + cg);
        Wts[(cg + 0) * 36 + k] = (_Float16)w.x;
        Wts[(cg + 1) * 36 + k] = (_Float16)w.y;
        Wts[(cg + 2) * 36 + k] = (_Float16)w.z;
        Wts[(cg + 3) * 36 + k] = (_Float16)w.w;
      }
    }
    if (kt + 32 < DM) {  // L2 prefetch of next tiles (global_prefetch_b8)
      __builtin_prefetch(X + (size_t)(row0 + (t & 63)) * DM + kt + 32, 0, 1);
      __builtin_prefetch(W + (size_t)(kt + 32 + (t >> 2)) * DM + col0, 0, 1);
    }
    __syncthreads();
    const v16h a = load_frag_a(&Xs[wave * 16 * 36], 36);
#pragma unroll
    for (int nt = 0; nt < 4; ++nt) {
      const v16h b = load_frag_b(&Wts[nt * 16 * 36], 36);
      acc[nt] = wmma_f16(a, b, acc[nt]);
    }
  }

  const int lane = t & 31;
  const int n = lane & 15, mb = (lane >> 4) * 8;
#pragma unroll
  for (int nt = 0; nt < 4; ++nt) {
    const int gc = col0 + nt * 16 + n;
    const int h = gc >> 6, d = gc & 63;
#pragma unroll
    for (int r = 0; r < 8; ++r) {
      const int gr = row0 + wave * 16 + mb + r;
      const int b = gr >> 10, s = gr & 1023;
      O[(((size_t)b * NH + h) * SEQ + s) * DK + d] = (_Float16)acc[nt][r];
    }
  }
}

// ---------------------------------------------------------------------------
// Kernel 2: attention.  One workgroup per (b,h, 64-row q tile), 4 waves.
// K/V streamed through LDS in 128-column chunks (K via async global->LDS
// DMA when available); 3 passes (max, sum, probs + context accumulate).
// Static LDS = 48KB.
// ---------------------------------------------------------------------------
__global__ __launch_bounds__(128) void attn_kernel(
    const _Float16* __restrict__ Qh, const _Float16* __restrict__ Kh,
    const _Float16* __restrict__ Vh, const unsigned char* __restrict__ mask,
    float* __restrict__ attn_out, _Float16* __restrict__ ctx_out) {
  __shared__ _Float16 Qs[64 * 72];    // 9216 B
  __shared__ _Float16 Ks[128 * 72];   // 18432 B  (chunk of K, row-major s x d)
  __shared__ _Float16 VTs[64 * 136];  // 17408 B  (chunk of V, transposed d x s)
  __shared__ _Float16 Ps[4 * 16 * 32];// 4096 B   (per-wave prob patch)

  const int t = threadIdx.x, wave = t >> 5, lane = t & 31;
  const int n = lane & 15, mb = (lane >> 4) * 8;
  const int bh = blockIdx.y, b = bh >> 4, h = bh & 15;
  const int q0 = blockIdx.x * 64;

  const _Float16* Qg = Qh + ((size_t)bh * SEQ + q0) * DK;
  const _Float16* Kg = Kh + (size_t)bh * SEQ * DK;
  const _Float16* Vg = Vh + (size_t)bh * SEQ * DK;

  // ---- load Q tile (64x64 f16) once ----
#pragma unroll
  for (int i = 0; i < 4; ++i) {
    const int c = t + i * 128, row = c >> 3, cg = (c & 7) * 8;
    copy16_g2l(Qg + row * DK + cg, &Qs[row * 72 + cg]);
  }
  async_wait();
  __syncthreads();
  const v16h qa0 = load_frag_a(&Qs[wave * 16 * 72], 72);       // d 0..31
  const v16h qa1 = load_frag_a(&Qs[wave * 16 * 72 + 32], 72);  // d 32..63

  auto load_k_chunk = [&](int kb) {
    const _Float16* src = Kg + (size_t)kb * 128 * DK;
#pragma unroll
    for (int i = 0; i < 8; ++i) {
      const int c = t + i * 128, row = c >> 3, cg = (c & 7) * 8;
      copy16_g2l(src + row * DK + cg, &Ks[row * 72 + cg]);
    }
  };
  auto load_v_chunk = [&](int kb) {  // transposed: needs VGPR round-trip
    const _Float16* src = Vg + (size_t)kb * 128 * DK;
#pragma unroll
    for (int i = 0; i < 8; ++i) {
      const int c = t + i * 128, row = c >> 3, cg = (c & 7) * 8;
      uint4 vv = *(const uint4*)(src + row * DK + cg);
      const _Float16* hp = (const _Float16*)&vv;
#pragma unroll
      for (int j = 0; j < 8; ++j) VTs[(cg + j) * 136 + row] = hp[j];
    }
  };

  // scores tile (16q x 16k): Q[16x64] . K[16x64]^T, scaled + masked
  auto score_tile = [&](int ktl, int kb) -> v8f {
    v8f c = {};
    const v16h b0 = load_frag_b(&Ks[(ktl * 16) * 72], 72);       // d 0..31
    const v16h b1 = load_frag_b(&Ks[(ktl * 16) * 72 + 32], 72);  // d 32..63
    c = wmma_f16(qa0, b0, c);
    c = wmma_f16(qa1, b1, c);
    const int kcol = kb * 128 + ktl * 16 + n;
#pragma unroll
    for (int r = 0; r < 8; ++r) {
      const int q = q0 + wave * 16 + mb + r;
      float s = c[r] * 0.125f;  // 1/sqrt(64)
      if (mask[((size_t)b * SEQ + q) * SEQ + kcol]) s = -1e9f;
      c[r] = s;
    }
    return c;
  };

  // ---- pass A: row max ----
  float rmax[8];
#pragma unroll
  for (int r = 0; r < 8; ++r) rmax[r] = -3.0e38f;
  for (int kb = 0; kb < 8; ++kb) {
    __syncthreads();
    load_k_chunk(kb);
    async_wait();
    __syncthreads();
    for (int ktl = 0; ktl < 8; ++ktl) {
      const v8f c = score_tile(ktl, kb);
#pragma unroll
      for (int r = 0; r < 8; ++r) rmax[r] = fmaxf(rmax[r], c[r]);
    }
  }
#pragma unroll
  for (int m = 1; m < 16; m <<= 1)
#pragma unroll
    for (int r = 0; r < 8; ++r)
      rmax[r] = fmaxf(rmax[r], __shfl_xor(rmax[r], m, 32));

  // ---- pass B: row sum of exp ----
  float rsum[8] = {0.f, 0.f, 0.f, 0.f, 0.f, 0.f, 0.f, 0.f};
  for (int kb = 0; kb < 8; ++kb) {
    __syncthreads();
    load_k_chunk(kb);
    async_wait();
    __syncthreads();
    for (int ktl = 0; ktl < 8; ++ktl) {
      const v8f c = score_tile(ktl, kb);
#pragma unroll
      for (int r = 0; r < 8; ++r) rsum[r] += expf(c[r] - rmax[r]);
    }
  }
#pragma unroll
  for (int m = 1; m < 16; m <<= 1)
#pragma unroll
    for (int r = 0; r < 8; ++r) rsum[r] += __shfl_xor(rsum[r], m, 32);
  float rinv[8];
#pragma unroll
  for (int r = 0; r < 8; ++r) rinv[r] = 1.0f / rsum[r];

  // ---- pass C: probs out + context = P.V ----
  v8f cacc[4] = {};
  _Float16* Pw = &Ps[wave * 16 * 32];
  for (int kb = 0; kb < 8; ++kb) {
    __syncthreads();
    load_k_chunk(kb);
    load_v_chunk(kb);
    async_wait();
    __syncthreads();
    for (int kp = 0; kp < 8; kp += 2) {
#pragma unroll
      for (int u = 0; u < 2; ++u) {
        const v8f c = score_tile(kp + u, kb);
        const int kcol = kb * 128 + (kp + u) * 16 + n;
#pragma unroll
        for (int r = 0; r < 8; ++r) {
          const float p = expf(c[r] - rmax[r]) * rinv[r];
          const int q = q0 + wave * 16 + mb + r;
          attn_out[((size_t)bh * SEQ + q) * SEQ + kcol] = p;
          Pw[(mb + r) * 32 + u * 16 + n] = (_Float16)p;  // A-frag repack
        }
      }
      const v16h pa = load_frag_a(Pw, 32);  // 16q x 32k probs
#pragma unroll
      for (int dt = 0; dt < 4; ++dt) {
        // B[k][d] = V[kb*128+kp*16+k][dt*16+d] = VTs[(dt*16+d)*136 + kp*16+k]
        const v16h vb = load_frag_b(&VTs[(dt * 16) * 136 + kp * 16], 136);
        cacc[dt] = wmma_f16(pa, vb, cacc[dt]);
      }
    }
  }

  // ---- store context f16 as [B,S,H*64] ----
#pragma unroll
  for (int dt = 0; dt < 4; ++dt) {
#pragma unroll
    for (int r = 0; r < 8; ++r) {
      const int q = q0 + wave * 16 + mb + r;
      ctx_out[((size_t)b * SEQ + q) * DM + h * DK + dt * 16 + n] =
          (_Float16)cacc[dt][r];
    }
  }
}

// ---------------------------------------------------------------------------
// Kernel 3a: output projection + residual.  ctx(f16)[8192,1024] @ Wo(f32)
// + input_Q -> tmp(f32).  Same tiling as kernel 1.
// ---------------------------------------------------------------------------
__global__ __launch_bounds__(128) void out_proj_kernel(
    const _Float16* __restrict__ Ctx, const float* __restrict__ Wo,
    const float* __restrict__ resid, float* __restrict__ Tmp) {
  __shared__ _Float16 Xs[64 * 36];
  __shared__ _Float16 Wts[64 * 36];

  const int t = threadIdx.x, wave = t >> 5;
  const int row0 = blockIdx.x * 64, col0 = blockIdx.y * 64;
  v8f acc[4] = {};

  for (int kt = 0; kt < DM; kt += 32) {
    __syncthreads();
    {  // ctx tile already f16
      const int r = t >> 3, cg = (t & 7) * 4;
#pragma unroll
      for (int i = 0; i < 4; ++i) {
        const int row = r + i * 16;
        *(uint2*)&Xs[row * 36 + cg] =
            *(const uint2*)(Ctx + (size_t)(row0 + row) * DM + kt + cg);
      }
    }
    {  // Wo tile f32 -> f16, transposed
      const int r = t >> 4, cg = (t & 15) * 4;
#pragma unroll
      for (int i = 0; i < 4; ++i) {
        const int k = r + i * 8;
        const float4 w =
            *(const float4*)(Wo + (size_t)(kt + k) * DM + col0 + cg);
        Wts[(cg + 0) * 36 + k] = (_Float16)w.x;
        Wts[(cg + 1) * 36 + k] = (_Float16)w.y;
        Wts[(cg + 2) * 36 + k] = (_Float16)w.z;
        Wts[(cg + 3) * 36 + k] = (_Float16)w.w;
      }
    }
    if (kt + 32 < DM) {
      __builtin_prefetch(Ctx + (size_t)(row0 + (t & 63)) * DM + kt + 32, 0, 1);
      __builtin_prefetch(Wo + (size_t)(kt + 32 + (t >> 2)) * DM + col0, 0, 1);
    }
    __syncthreads();
    const v16h a = load_frag_a(&Xs[wave * 16 * 36], 36);
#pragma unroll
    for (int nt = 0; nt < 4; ++nt) {
      const v16h b = load_frag_b(&Wts[nt * 16 * 36], 36);
      acc[nt] = wmma_f16(a, b, acc[nt]);
    }
  }

  const int lane = t & 31;
  const int n = lane & 15, mb = (lane >> 4) * 8;
#pragma unroll
  for (int nt = 0; nt < 4; ++nt) {
#pragma unroll
    for (int r = 0; r < 8; ++r) {
      const int gr = row0 + wave * 16 + mb + r;
      const int gc = col0 + nt * 16 + n;
      Tmp[(size_t)gr * DM + gc] = acc[nt][r] + resid[(size_t)gr * DM + gc];
    }
  }
}

// ---------------------------------------------------------------------------
// Kernel 3b: row-wise LayerNorm (gamma=1, beta=0), 1024 elems/row.
// ---------------------------------------------------------------------------
__global__ __launch_bounds__(256) void ln_kernel(const float* __restrict__ x,
                                                 float* __restrict__ out) {
  __shared__ float ss[8], ss2[8];
  const int row = blockIdx.x;
  const float* p = x + (size_t)row * DM;
  float s = 0.f, s2 = 0.f;
  for (int i = threadIdx.x; i < DM; i += 256) {
    const float v = p[i];
    s += v;
    s2 += v * v;
  }
#pragma unroll
  for (int m = 1; m < 32; m <<= 1) {
    s += __shfl_xor(s, m, 32);
    s2 += __shfl_xor(s2, m, 32);
  }
  const int wave = threadIdx.x >> 5, lane = threadIdx.x & 31;
  if (lane == 0) { ss[wave] = s; ss2[wave] = s2; }
  __syncthreads();
  float ts = 0.f, ts2 = 0.f;
#pragma unroll
  for (int i = 0; i < 8; ++i) { ts += ss[i]; ts2 += ss2[i]; }
  const float mean = ts * (1.0f / DM);
  const float var = ts2 * (1.0f / DM) - mean * mean;
  const float rstd = rsqrtf(var + 1e-5f);
  for (int i = threadIdx.x; i < DM; i += 256)
    out[(size_t)row * DM + i] = (p[i] - mean) * rstd;
}

// ---------------------------------------------------------------------------
extern "C" void kernel_launch(void* const* d_in, const int* in_sizes, int n_in,
                              void* d_out, int out_size, void* d_ws,
                              size_t ws_size, hipStream_t stream) {
  (void)in_sizes; (void)n_in; (void)out_size; (void)ws_size;
  const float* inQ = (const float*)d_in[0];
  const float* inK = (const float*)d_in[1];
  const float* inV = (const float*)d_in[2];
  const unsigned char* mask = (const unsigned char*)d_in[3];
  const float* Wq = (const float*)d_in[4];
  const float* Wk = (const float*)d_in[5];
  const float* Wv = (const float*)d_in[6];
  const float* Wo = (const float*)d_in[7];

  char* ws = (char*)d_ws;
  _Float16* Qh  = (_Float16*)(ws);                       // 16 MB [B,H,S,64]
  _Float16* Kh  = (_Float16*)(ws + ((size_t)16 << 20));  // 16 MB
  _Float16* Vh  = (_Float16*)(ws + ((size_t)32 << 20));  // 16 MB
  _Float16* Ctx = (_Float16*)(ws + ((size_t)48 << 20));  // 16 MB [B,S,H*64]
  float*    Tmp = (float*)(ws);  // 32 MB, reuses Qh/Kh (dead by then)

  float* out  = (float*)d_out;                       // 8*1024*1024
  float* attn = out + (size_t)BATCH * SEQ * DM;      // 8*16*1024*1024

  qkv_proj_kernel<<<dim3(128, 16, 3), 128, 0, stream>>>(inQ, inK, inV, Wq, Wk,
                                                        Wv, Qh, Kh, Vh);
  attn_kernel<<<dim3(16, 128), 128, 0, stream>>>(Qh, Kh, Vh, mask, attn, Ctx);
  out_proj_kernel<<<dim3(128, 16), 128, 0, stream>>>(Ctx, Wo, inQ, Tmp);
  ln_kernel<<<dim3(BATCH * SEQ), 256, 0, stream>>>(Tmp, out);
}
